// MultiHeadAttention_15066745274737
// MI455X (gfx1250) — matrix operand order repaired and checked by reading, not compiled, not run on hardware
//
#include <hip/hip_runtime.h>

typedef __bf16 bf16;
typedef __attribute__((ext_vector_type(16))) __bf16 v16bf;
typedef __attribute__((ext_vector_type(8)))  float  v8f;
typedef __attribute__((ext_vector_type(4)))  unsigned v4u;
typedef __attribute__((ext_vector_type(8)))  int      v8i;
typedef __attribute__((ext_vector_type(4)))  int      v4i;

#define DEV __device__ __forceinline__

constexpr int B_ = 2;
constexpr int T_ = 2048;
constexpr int S_ = 2048;
constexpr int E_ = 1024;
constexpr int H_ = 16;
constexpr int Dh = 64;

// ---------------- WMMA fragment helpers (ISA 7.12.2 layouts) ----------------
DEV v16bf frag_a(const bf16* base, int ld, int koff, int lane) {
  int r  = lane & 15;
  int hi = lane >> 4;
  union { v16bf v; unsigned u[8]; } f;
  const bf16* p = base + r * ld + koff + hi * 8;
#pragma unroll
  for (int i = 0; i < 8; ++i) {
    int k = (i < 4) ? (2 * i) : (16 + 2 * (i - 4));
    f.u[i] = *(const unsigned*)(p + k);
  }
  return f.v;
}

DEV v16bf frag_b(const bf16* base, int ld, int koff, int lane) {
  int n  = lane & 15;
  int hi = lane >> 4;
  union { v16bf v; unsigned u[8]; } f;
  const bf16* p = base + n * ld + koff + hi * 8;
#pragma unroll
  for (int i = 0; i < 8; ++i)
    f.u[i] = *(const unsigned*)(p + ((i < 4) ? (2 * i) : (16 + 2 * (i - 4))));
  return f.v;
}

DEV v8f wmma_bf16(v16bf a, v16bf b, v8f c) {
  return __builtin_amdgcn_wmma_f32_16x16x32_bf16(false, a, false, b,
                                                 (short)0, c, false, false);
}

// ---------------- Tensor Data Mover (async 2D tile -> LDS) ------------------
// D# per ISA ch.8: group0 = {count, lds_addr, global_addr, type=2};
// group1 = {data_size=2B, pad_enable, pad_interval, pad_amount=4 dwords,
//           tensor_dim0/1, tile_dim0/1, tensor_dim0_stride}.
// pad turns a w-element bf16 row into an LDS stride of w+8 elements.
#if __has_builtin(__builtin_amdgcn_tensor_load_to_lds) && \
    __has_builtin(__builtin_amdgcn_s_wait_tensorcnt)
#define USE_TDM 1
#endif

DEV void tdm_load_2d(bf16* lds, const bf16* g, unsigned w, unsigned rows,
                     unsigned stride, unsigned ldst, unsigned padiv, int lane) {
#ifdef USE_TDM
  (void)ldst; (void)lane;
  unsigned long long ga = (unsigned long long)(uintptr_t)g;
  v4u g0;
  g0.x = 1u;                                        // count = 1 descriptor
  g0.y = (unsigned)(uintptr_t)lds;                  // lds_addr (offset bits)
  g0.z = (unsigned)ga;                              // global_addr[31:0]
  g0.w = (unsigned)((ga >> 32) & 0x01FFFFFFu) | (2u << 30);  // addr[56:32]|type=2
  v8i g1;
  g1[0] = (1 << 16)            // data_size = 2 bytes
        | (1 << 20)            // pad_enable
        | ((int)padiv << 22)   // pad_interval (4 -> 128B rows, 5 -> 256B rows)
        | (3 << 25);           // pad_amount = 4 dwords (16B) -> +8 bf16 skew
  g1[1] = (int)(w << 16);      // tensor_dim0  (bits 79:48)
  g1[2] = (int)(rows << 16);   // tensor_dim1  (bits 111:80)
  g1[3] = (int)(w << 16);      // tile_dim0    (bits 127:112)
  g1[4] = (int)rows;           // tile_dim1    (bits 143:128)
  g1[5] = (int)stride;         // tensor_dim0_stride (bits 207:160)
  g1[6] = 0;
  g1[7] = 0;
  v4i z4 = (v4i)0;
  v8i z8 = (v8i)0;
  __builtin_amdgcn_tensor_load_to_lds(g0, g1, z4, z4, z8, 0);
#else
  for (unsigned i = (unsigned)lane; i < w * rows; i += 32) {
    unsigned r = i / w, c = i % w;
    lds[r * ldst + c] = g[(size_t)r * stride + c];
  }
#endif
}

DEV void tdm_wait() {
#ifdef USE_TDM
  __builtin_amdgcn_s_wait_tensorcnt(0);
#endif
}

// ---------------- Projection / output GEMM: Y = X @ W^T + bias -------------
// MODE 0: X fp32 -> bf16 head-split [(b*H+h)][row][d], scaled   (Q, K)
// MODE 2: X fp32 -> bf16 transposed [(b*H+h)][d][row]           (V)
// MODE 1: X bf16 (ctx) -> fp32 plain [m][n]                      (out proj)
template <int MODE>
__global__ __launch_bounds__(256) void gemm128(
    const float* __restrict__ Xf, const bf16* __restrict__ Xb,
    const float* __restrict__ W,  const float* __restrict__ bias,
    float scale, bf16* __restrict__ Yb, float* __restrict__ Yf) {
  __shared__ __align__(16) bf16 sA[128 * 40];
  __shared__ __align__(16) bf16 sB[128 * 40];
  const int tid = threadIdx.x;
  const int lane = tid & 31, w = tid >> 5;
  const int wm = w & 3, wn = w >> 2;           // 4x2 wave grid, 32x64 per wave
  const int m0 = blockIdx.x * 128, n0 = blockIdx.y * 128;

  v8f c[2][4];
#pragma unroll
  for (int i = 0; i < 2; ++i)
#pragma unroll
    for (int j = 0; j < 4; ++j) c[i][j] = (v8f)(0.0f);

  for (int k0 = 0; k0 < 1024; k0 += 32) {
    __syncthreads();
    for (int i = tid; i < 128 * 32; i += 256) {
      int r = i >> 5, kk = i & 31;
      float x;
      if (MODE == 1) x = (float)Xb[(size_t)(m0 + r) * 1024 + k0 + kk];
      else           x = Xf[(size_t)(m0 + r) * 1024 + k0 + kk];
      sA[r * 40 + kk] = (bf16)x;
    }
    for (int i = tid; i < 128 * 32; i += 256) {
      int n = i >> 5, kk = i & 31;
      sB[n * 40 + kk] = (bf16)W[(size_t)(n0 + n) * 1024 + k0 + kk];
    }
    if (tid < 128 && k0 + 32 < 1024) {           // prefetch next k-slab
      if (MODE == 1) __builtin_prefetch(&Xb[(size_t)(m0 + tid) * 1024 + k0 + 32]);
      else           __builtin_prefetch(&Xf[(size_t)(m0 + tid) * 1024 + k0 + 32]);
      __builtin_prefetch(&W[(size_t)(n0 + tid) * 1024 + k0 + 32]);
    }
    __syncthreads();

    v16bf a0 = frag_a(sA + (wm * 32) * 40, 40, 0, lane);
    v16bf a1 = frag_a(sA + (wm * 32 + 16) * 40, 40, 0, lane);
#pragma unroll
    for (int j = 0; j < 4; ++j) {
      v16bf b = frag_b(sB + (wn * 64 + j * 16) * 40, 40, 0, lane);
      c[0][j] = wmma_bf16(a0, b, c[0][j]);
      c[1][j] = wmma_bf16(a1, b, c[1][j]);
    }
  }

#pragma unroll
  for (int i = 0; i < 2; ++i)
#pragma unroll
    for (int j = 0; j < 4; ++j)
#pragma unroll
      for (int r = 0; r < 8; ++r) {
        int M = m0 + wm * 32 + i * 16 + r + (lane >> 4) * 8;
        int N = n0 + wn * 64 + j * 16 + (lane & 15);
        float v = (c[i][j][r] + bias[N]) * scale;
        int bb = M >> 11, row = M & 2047, h = N >> 6, d = N & 63;
        if (MODE == 0)
          Yb[(((size_t)(bb * H_ + h)) * T_ + row) * Dh + d] = (bf16)v;
        else if (MODE == 2)
          Yb[(((size_t)(bb * H_ + h)) * Dh + d) * S_ + row] = (bf16)v;
        else
          Yf[(size_t)M * 1024 + N] = v;
      }
}

// ---------------- Fused attention ------------------------------------------
// One block = one (b,h) x 128 query rows. 8 waves, each owns 16 rows x the
// full 128-key S-tile (softmax reduction stays inside one wave). K and V^T
// tiles are staged by the Tensor Data Mover, double-buffered against compute.
__global__ __launch_bounds__(256) void attn_kernel(
    const bf16* __restrict__ Qp, const bf16* __restrict__ Kp,
    const bf16* __restrict__ Vt, const int* __restrict__ mask,
    float* __restrict__ attn, bf16* __restrict__ ctx) {
  __shared__ __align__(16) bf16 sQ[128 * 72];
  __shared__ __align__(16) bf16 sK[2][128 * 72];
  __shared__ __align__(16) bf16 sVT[2][64 * 136];  // [d][key], skewed
  __shared__ __align__(16) bf16 sP[128 * 136];     // probs [row][key]
  __shared__ float sM[2][128];

  const int tid = threadIdx.x, lane = tid & 31, w = tid >> 5;
  const int tt = blockIdx.x, bh = blockIdx.y;
  const int b = bh >> 4, h = bh & 15;
  const int t0 = tt * 128;
  const bf16* Qg = Qp + ((size_t)bh * T_ + t0) * Dh;
  const bf16* Kg = Kp + (size_t)bh * S_ * Dh;
  const bf16* Vg = Vt + (size_t)bh * Dh * S_;      // [d][s]
  const int*  mg = mask + (size_t)b * S_;

  // ---- prologue: TDM Q tile + K tile 0, mask tile 0 ----
  if (tid < 32) {
    tdm_load_2d(sQ,    Qg, 64, 128, 64, 72, 4, lane);
    tdm_load_2d(sK[0], Kg, 64, 128, 64, 72, 4, lane);
  }
  for (int i = tid; i < 128; i += 256) {
    int mv = mg[i];
    sM[0][i] = mv ? (float)mv : -9.0e15f;
  }
  if (tid < 32) tdm_wait();
  __syncthreads();

  v16bf qa0 = frag_a(sQ + (w * 16) * 72, 72, 0,  lane);
  v16bf qa1 = frag_a(sQ + (w * 16) * 72, 72, 32, lane);

  float mrow[8], lrow[8];
#pragma unroll
  for (int r = 0; r < 8; ++r) { mrow[r] = -3.0e38f; lrow[r] = 0.0f; }

  // ---- Pass A: online row max / sumexp ----
  for (int st = 0; st < S_ / 128; ++st) {
    const int cur = st & 1, nxt = cur ^ 1;
    if (st + 1 < S_ / 128) {
      if (tid < 32)
        tdm_load_2d(sK[nxt], Kg + (size_t)(st + 1) * 128 * 64, 64, 128, 64, 72, 4, lane);
      for (int i = tid; i < 128; i += 256) {
        int mv = mg[(st + 1) * 128 + i];
        sM[nxt][i] = mv ? (float)mv : -9.0e15f;
      }
    }

    v8f c[8];
#pragma unroll
    for (int ct = 0; ct < 8; ++ct) c[ct] = (v8f)(0.0f);
#pragma unroll
    for (int ct = 0; ct < 8; ++ct) {
      v16bf b0 = frag_b(sK[cur] + (ct * 16) * 72, 72, 0,  lane);
      v16bf b1 = frag_b(sK[cur] + (ct * 16) * 72, 72, 32, lane);
      c[ct] = wmma_bf16(qa0, b0, c[ct]);
      c[ct] = wmma_bf16(qa1, b1, c[ct]);
    }
#pragma unroll
    for (int r = 0; r < 8; ++r) {
      float tmax = -3.0e38f;
#pragma unroll
      for (int ct = 0; ct < 8; ++ct) {
        float x = c[ct][r];
        x = fminf(fmaxf(x, -50000.0f), 50000.0f);
        x += sM[cur][ct * 16 + (lane & 15)];
        c[ct][r] = x;
        tmax = fmaxf(tmax, x);
      }
#pragma unroll
      for (int off = 1; off < 16; off <<= 1)
        tmax = fmaxf(tmax, __shfl_xor(tmax, off, 32));
      float mnew = fmaxf(mrow[r], tmax);
      float sum = 0.0f;
#pragma unroll
      for (int ct = 0; ct < 8; ++ct) sum += __expf(c[ct][r] - mnew);
#pragma unroll
      for (int off = 1; off < 16; off <<= 1)
        sum += __shfl_xor(sum, off, 32);
      lrow[r] = lrow[r] * __expf(mrow[r] - mnew) + sum;
      mrow[r] = mnew;
    }
    if (tid < 32 && st + 1 < S_ / 128) tdm_wait();
    __syncthreads();
  }

  float rl[8];
#pragma unroll
  for (int r = 0; r < 8; ++r) rl[r] = (lrow[r] > 0.0f) ? 1.0f / lrow[r] : 0.0f;

  v8f o[4];
#pragma unroll
  for (int nt = 0; nt < 4; ++nt) o[nt] = (v8f)(0.0f);

  // ---- Pass B prologue: K tile 0 + V^T tile 0 ----
  if (tid < 32) {
    tdm_load_2d(sK[0],  Kg, 64, 128, 64,   72,  4, lane);
    tdm_load_2d(sVT[0], Vg, 128, 64, 2048, 136, 5, lane);
  }
  for (int i = tid; i < 128; i += 256) {
    int mv = mg[i];
    sM[0][i] = mv ? (float)mv : -9.0e15f;
  }
  if (tid < 32) tdm_wait();
  __syncthreads();

  // ---- Pass B: recompute S, normalize, emit probs, accumulate O = P @ V ----
  for (int st = 0; st < S_ / 128; ++st) {
    const int cur = st & 1, nxt = cur ^ 1;
    if (st + 1 < S_ / 128) {
      if (tid < 32) {
        tdm_load_2d(sK[nxt],  Kg + (size_t)(st + 1) * 128 * 64, 64, 128, 64, 72, 4, lane);
        tdm_load_2d(sVT[nxt], Vg + (size_t)(st + 1) * 128, 128, 64, 2048, 136, 5, lane);
      }
      for (int i = tid; i < 128; i += 256) {
        int mv = mg[(st + 1) * 128 + i];
        sM[nxt][i] = mv ? (float)mv : -9.0e15f;
      }
    }

    v8f c[8];
#pragma unroll
    for (int ct = 0; ct < 8; ++ct) c[ct] = (v8f)(0.0f);
#pragma unroll
    for (int ct = 0; ct < 8; ++ct) {
      v16bf b0 = frag_b(sK[cur] + (ct * 16) * 72, 72, 0,  lane);
      v16bf b1 = frag_b(sK[cur] + (ct * 16) * 72, 72, 32, lane);
      c[ct] = wmma_bf16(qa0, b0, c[ct]);
      c[ct] = wmma_bf16(qa1, b1, c[ct]);
    }
#pragma unroll
    for (int r = 0; r < 8; ++r) {
      int Mrow = w * 16 + r + (lane >> 4) * 8;
      size_t arow = ((size_t)bh * T_ + (t0 + Mrow)) * S_ + (size_t)st * 128;
#pragma unroll
      for (int ct = 0; ct < 8; ++ct) {
        float x = c[ct][r];
        x = fminf(fmaxf(x, -50000.0f), 50000.0f);
        x += sM[cur][ct * 16 + (lane & 15)];
        float p = __expf(x - mrow[r]) * rl[r];
        int col = ct * 16 + (lane & 15);
        attn[arow + col] = p;                    // required attn output
        sP[Mrow * 136 + col] = (bf16)p;          // stage for P@V
      }
    }
    __syncthreads();                             // sP ready

#pragma unroll
    for (int ks = 0; ks < 4; ++ks) {
      v16bf pa = frag_a(sP + (w * 16) * 136, 136, ks * 32, lane);
#pragma unroll
      for (int nt = 0; nt < 4; ++nt) {
        v16bf vb = frag_b(sVT[cur] + (nt * 16) * 136, 136, ks * 32, lane);
        o[nt] = wmma_bf16(pa, vb, o[nt]);
      }
    }
    if (tid < 32 && st + 1 < S_ / 128) tdm_wait();
    __syncthreads();
  }

  // ctx layout [b][t][h*64+d] (bthd), bf16 for the output projection
#pragma unroll
  for (int nt = 0; nt < 4; ++nt)
#pragma unroll
    for (int r = 0; r < 8; ++r) {
      int Mrow = t0 + w * 16 + r + (lane >> 4) * 8;
      int d = nt * 16 + (lane & 15);
      ctx[((size_t)b * T_ + Mrow) * E_ + h * 64 + d] = (bf16)o[nt][r];
    }
}

// ---------------------------------------------------------------------------
extern "C" void kernel_launch(void* const* d_in, const int* in_sizes, int n_in,
                              void* d_out, int out_size, void* d_ws, size_t ws_size,
                              hipStream_t stream) {
  (void)in_sizes; (void)n_in; (void)out_size; (void)ws_size;
  const float* q  = (const float*)d_in[0];
  const float* k  = (const float*)d_in[1];
  const float* v  = (const float*)d_in[2];
  const int*   am = (const int*)d_in[3];
  const float* Wq = (const float*)d_in[4];  const float* bq = (const float*)d_in[5];
  const float* Wk = (const float*)d_in[6];  const float* bk = (const float*)d_in[7];
  const float* Wv = (const float*)d_in[8];  const float* bv = (const float*)d_in[9];
  const float* Wo = (const float*)d_in[10]; const float* bo = (const float*)d_in[11];

  float* out  = (float*)d_out;
  float* attn = out + (size_t)B_ * T_ * E_;          // 4,194,304 floats in

  const size_t per = (size_t)B_ * H_ * T_ * Dh;      // 4,194,304 bf16 elems
  bf16* Qp  = (bf16*)d_ws;
  bf16* Kp  = Qp + per;
  bf16* Vp  = Kp + per;                              // [bh][d][s] (transposed)
  bf16* Ctx = Vp + per;                              // [b][t][E] bf16

  dim3 blk(256);
  dim3 gridP(32, 8);                                 // 4096/128 x 1024/128
  const float scale = 0.125f;                        // 64^-0.5

  gemm128<0><<<gridP, blk, 0, stream>>>(q, nullptr, Wq, bq, scale, Qp, nullptr);
  gemm128<0><<<gridP, blk, 0, stream>>>(k, nullptr, Wk, bk, 1.0f,  Kp, nullptr);
  gemm128<2><<<gridP, blk, 0, stream>>>(v, nullptr, Wv, bv, 1.0f,  Vp, nullptr);

  attn_kernel<<<dim3(T_ / 128, B_ * H_), blk, 0, stream>>>(Qp, Kp, Vp, am, attn, Ctx);

  gemm128<1><<<gridP, blk, 0, stream>>>(nullptr, Ctx, Wo, bo, 1.0f, nullptr, out);
}
